// SimpleTensorAggregateLayer_7103875908194
// MI455X (gfx1250) — compile-verified, matrix operand order checked by reading, not run on hardware
//
#include <hip/hip_runtime.h>
#include <hip/hip_bf16.h>

typedef __attribute__((ext_vector_type(16))) _Float16 v16h;
typedef __attribute__((ext_vector_type(8)))  float    v8f;
typedef __attribute__((ext_vector_type(2)))  float    v2f;

#if __has_builtin(__builtin_amdgcn_wmma_f32_16x16x4_f32)
#define STAL_USE_F32_WMMA 1
#else
#define STAL_USE_F32_WMMA 0
#endif

namespace {
constexpr float kCutoff  = 5.0f;
constexpr float kInvNorm = 1.0f / 16.0f;   // 1 / NORM_FACTOR
constexpr float kPi      = 3.14159265358979323846f;
}

// One wave32 per 16-edge tile.
//  Stage 1: build rbf operands and run WMMA -> fn0/fn1/fn2 for 16 edges x 16 ch.
//           Preferred path: 4 chained v_wmma_f32_16x16x4_f32 per weight matrix
//           (full f32, exact-precision match to the reference einsum).
//           Fallback: one v_wmma_f32_16x16x32_f16 per weight matrix (K padded).
//  Stage 2: each lane walks its 8 (edge, channel) cells of the D layout,
//           gathers T0/T1/T2[idx_j], does the equivariant contractions, and
//           atomically scatters 13 floats into out[idx_i].
__global__ __launch_bounds__(256) void stal_fused_kernel(
    const float* __restrict__ coord,
    const float* __restrict__ T0,
    const float* __restrict__ T1,
    const float* __restrict__ T2,
    const float* __restrict__ W0,
    const float* __restrict__ W1,
    const float* __restrict__ W2,
    const int*   __restrict__ eidx_i,
    const int*   __restrict__ eidx_j,
    float* __restrict__ out0,
    float* __restrict__ out1,
    float* __restrict__ out2,
    int n_edges)
{
  const int lane = threadIdx.x & 31;
  const int wave = threadIdx.x >> 5;
  const int tile = blockIdx.x * (blockDim.x >> 5) + wave;
  const int n_tiles = (n_edges + 15) >> 4;
  if (tile >= n_tiles) return;          // wave-uniform: EXEC stays all-1s
  const int ebase = tile << 4;
  const int hi = lane >> 4;             // half-wave select
  const int lo = lane & 15;             // A-row / B-column / channel

  // ---------------- Stage 1: edge geometry for this lane's A-row ----------
  int erow = ebase + lo;
  if (erow >= n_edges) erow = n_edges - 1;   // clamp: result unused on tail
  float theta, invd;
  {
    const int ai = eidx_i[erow];
    const int aj = eidx_j[erow];
    const float rx = coord[3 * aj + 0] - coord[3 * ai + 0];
    const float ry = coord[3 * aj + 1] - coord[3 * ai + 1];
    const float rz = coord[3 * aj + 2] - coord[3 * ai + 2];
    const float d2 = rx * rx + ry * ry + rz * rz;
    invd = rsqrtf(d2);
    const float d = d2 * invd;
    theta = (kPi / kCutoff) * d;
  }

  v8f F0, F1, F2;

#if STAL_USE_F32_WMMA
  // ---- f32 path: D = A(16x4) x B(4x16) + C, chained over 4 K-steps. ----
  // A: lane l holds row M=l&15; element e <-> K = 2*hi + e (within step).
  // B: element e <-> K = 2*hi + e, N = lo.
  // Lane's rbf features across steps: f = 4*s + 2*hi + e.
  float rbf[8];
#pragma unroll
  for (int s = 0; s < 4; ++s) {
#pragma unroll
    for (int e = 0; e < 2; ++e) {
      const int f = 4 * s + 2 * hi + e;
      rbf[2 * s + e] = __sinf((float)(f + 1) * theta) * invd;
    }
  }
  F0 = (v8f){};
  F1 = (v8f){};
  F2 = (v8f){};
#pragma unroll
  for (int s = 0; s < 4; ++s) {
    v2f aa;
    aa[0] = rbf[2 * s + 0];
    aa[1] = rbf[2 * s + 1];
    const int fb = 4 * s + 2 * hi;
    v2f b0, b1, b2;
    b0[0] = W0[(fb + 0) * 16 + lo];  b0[1] = W0[(fb + 1) * 16 + lo];
    b1[0] = W1[(fb + 0) * 16 + lo];  b1[1] = W1[(fb + 1) * 16 + lo];
    b2[0] = W2[(fb + 0) * 16 + lo];  b2[1] = W2[(fb + 1) * 16 + lo];
    F0 = __builtin_amdgcn_wmma_f32_16x16x4_f32(false, aa, false, b0,
                                               (short)0, F0, false, false);
    F1 = __builtin_amdgcn_wmma_f32_16x16x4_f32(false, aa, false, b1,
                                               (short)0, F1, false, false);
    F2 = __builtin_amdgcn_wmma_f32_16x16x4_f32(false, aa, false, b2,
                                               (short)0, F2, false, false);
  }
#else
  // ---- f16 fallback: K padded 16 -> 32. ----
  // A: lanes 0-15 elems 0..7 = K0..7; lanes 16-31 elems 0..7 = K8..15.
  v16h A;
  {
    const int kbase = hi * 8;
#pragma unroll
    for (int e = 0; e < 8; ++e)
      A[e] = (_Float16)(__sinf((float)(kbase + e + 1) * theta) * invd);
#pragma unroll
    for (int e = 8; e < 16; ++e)
      A[e] = (_Float16)0.0f;
  }
  // B: element e <-> K = e + 16*hi, N = lo; K>=16 is the zero pad.
  v16h B0, B1, B2;
#pragma unroll
  for (int e = 0; e < 16; ++e) {
    float w0 = 0.0f, w1 = 0.0f, w2 = 0.0f;
    if (hi == 0) {
      w0 = W0[e * 16 + lo];
      w1 = W1[e * 16 + lo];
      w2 = W2[e * 16 + lo];
    }
    B0[e] = (_Float16)w0;
    B1[e] = (_Float16)w1;
    B2[e] = (_Float16)w2;
  }
  v8f zero = {};
  F0 = __builtin_amdgcn_wmma_f32_16x16x32_f16(false, A, false, B0,
                                              (short)0, zero, false, false);
  F1 = __builtin_amdgcn_wmma_f32_16x16x32_f16(false, A, false, B1,
                                              (short)0, zero, false, false);
  F2 = __builtin_amdgcn_wmma_f32_16x16x32_f16(false, A, false, B2,
                                              (short)0, zero, false, false);
#endif

  // ---------------- Stage 2: gather / contract / scatter ----------------
  // D layout: lane l, element r -> fn[edge ebase + r + 8*(l>>4)][channel l&15]
  const int c = lo;  // channel handled by this lane
#pragma unroll
  for (int r = 0; r < 8; ++r) {
    const int er = ebase + hi * 8 + r;
    const bool valid = er < n_edges;
    const int ec = valid ? er : 0;
    const int ii = eidx_i[ec];
    const int jj = eidx_j[ec];

    // unit bond vector (16 lanes of a half-wave share an edge -> broadcast)
    float ux = coord[3 * jj + 0] - coord[3 * ii + 0];
    float uy = coord[3 * jj + 1] - coord[3 * ii + 1];
    float uz = coord[3 * jj + 2] - coord[3 * ii + 2];
    const float qinv = rsqrtf(ux * ux + uy * uy + uz * uz);
    ux *= qinv; uy *= qinv; uz *= qinv;
    const float u[3] = {ux, uy, uz};

    // gather source tensors at (atom j, channel c) — coalesced per half-wave
    const float t0 = T0[jj * 16 + c];
    const float* t1p = T1 + (size_t)(jj * 16 + c) * 3;
    const float* t2p = T2 + (size_t)(jj * 16 + c) * 9;
    float t1[3], t2[9];
#pragma unroll
    for (int p = 0; p < 3; ++p) t1[p] = t1p[p];
#pragma unroll
    for (int p = 0; p < 9; ++p) t2[p] = t2p[p];

    const float f0 = F0[r], f1 = F1[r], f2 = F2[r];

    const float s1 = t1[0] * u[0] + t1[1] * u[1] + t1[2] * u[2];
    float v[3];
#pragma unroll
    for (int p = 0; p < 3; ++p)
      v[p] = t2[p * 3 + 0] * u[0] + t2[p * 3 + 1] * u[1] + t2[p * 3 + 2] * u[2];
    const float uv = u[0] * v[0] + u[1] * v[1] + u[2] * v[2];

    if (valid) {
      const float o0 = (t0 * f0 + f1 * s1 + f2 * uv) * kInvNorm;
      atomicAdd(&out0[ii * 16 + c], o0);

      const int b1 = (ii * 16 + c) * 3;
      const float g1 = t0 * f1 + f2 * s1;
#pragma unroll
      for (int p = 0; p < 3; ++p) {
        const float o1 = (t1[p] * f0 + g1 * u[p] + f1 * v[p]) * kInvNorm;
        atomicAdd(&out1[b1 + p], o1);
      }

      const int b2 = (ii * 16 + c) * 9;
#pragma unroll
      for (int p = 0; p < 3; ++p) {
        const float gp = f1 * t1[p] + f2 * (t0 * u[p] + v[p]);
#pragma unroll
        for (int q = 0; q < 3; ++q) {
          const float o2 = (t2[p * 3 + q] * f0 + gp * u[q]) * kInvNorm;
          atomicAdd(&out2[b2 + p * 3 + q], o2);
        }
      }
    }
  }
}

__global__ void stal_zero_kernel(float* __restrict__ p, int n) {
  const int i = blockIdx.x * blockDim.x + threadIdx.x;
  if (i < n) p[i] = 0.0f;
}

extern "C" void kernel_launch(void* const* d_in, const int* in_sizes, int n_in,
                              void* d_out, int out_size, void* d_ws, size_t ws_size,
                              hipStream_t stream) {
  const float* coord = (const float*)d_in[0];
  const float* T0    = (const float*)d_in[1];
  const float* T1    = (const float*)d_in[2];
  const float* T2    = (const float*)d_in[3];
  const float* W0    = (const float*)d_in[4];
  const float* W1    = (const float*)d_in[5];
  const float* W2    = (const float*)d_in[6];
  const int*   eidx  = (const int*)d_in[7];

  const int n_edges = in_sizes[7] / 2;
  const int n_atoms = in_sizes[0] / 3;

  float* out0 = (float*)d_out;                       // [A,16]
  float* out1 = out0 + (size_t)n_atoms * 16;         // [A,16,3]
  float* out2 = out1 + (size_t)n_atoms * 48;         // [A,16,3,3]

  // Outputs are accumulated atomically: re-zero every launch.
  {
    const int threads = 256;
    const int blocks = (out_size + threads - 1) / threads;
    stal_zero_kernel<<<blocks, threads, 0, stream>>>((float*)d_out, out_size);
  }

  const int n_tiles = (n_edges + 15) / 16;
  const int waves_per_block = 8;  // 256 threads = 8 wave32
  const int blocks = (n_tiles + waves_per_block - 1) / waves_per_block;
  stal_fused_kernel<<<blocks, 256, 0, stream>>>(
      coord, T0, T1, T2, W0, W1, W2,
      eidx, eidx + n_edges,
      out0, out1, out2, n_edges);
}
